// GraphAgent_42176578847132
// MI455X (gfx1250) — matrix operand order repaired and checked by reading, not compiled, not run on hardware
//
#include <hip/hip_runtime.h>
#include <hip/hip_bf16.h>

// ---------------------------------------------------------------------------
// Fused GNN (NNConv rank-1 edges + GRU, 6 steps) for MI455X (gfx1250).
// - One WG = 4 graphs (64 nodes): full recurrence in LDS, WMMA f16->f32 GEMMs.
// - Edge CSR binned by dst node (built per call) -> gather aggregation.
// ---------------------------------------------------------------------------

#define N_NODES 16384
#define N_EDGES 32768
#define N_GRAPHS 1024
#define N_STEMS 2048
#define OUT_PER_STEM 105
#define STEPS 6
#define SLOPE 0.01f
#define MAXE_WG 512   // edges cached per 64-node workgroup (expected ~128)

typedef __attribute__((ext_vector_type(16))) _Float16 v16h;
typedef __attribute__((ext_vector_type(8)))  float    v8f;

// ---- workspace layout -------------------------------------------------------
constexpr size_t A256(size_t x) { return (x + 255) & ~(size_t)255; }
constexpr size_t OFF_NODEH  = 0;                                   // 16384*64 f32
constexpr size_t OFF_CNT    = A256(OFF_NODEH + (size_t)N_NODES * 64 * 4);
constexpr size_t OFF_ROWPTR = A256(OFF_CNT + (size_t)N_NODES * 4);
constexpr size_t OFF_CURSOR = A256(OFF_ROWPTR + (size_t)(N_NODES + 1) * 4);
constexpr size_t OFF_META   = A256(OFF_CURSOR + (size_t)N_NODES * 4);
constexpr size_t OFF_WROOT  = A256(OFF_META + (size_t)N_EDGES * 4);
constexpr size_t OFF_WIH    = A256(OFF_WROOT + 64 * 64 * 2);
constexpr size_t OFF_WHH    = A256(OFF_WIH + 192 * 64 * 2);
constexpr size_t OFF_WB1    = A256(OFF_WHH + 192 * 64 * 2);
constexpr size_t OFF_WB2    = A256(OFF_WB1 + 64 * 96 * 2);
constexpr size_t OFF_WS1    = A256(OFF_WB2 + 64 * 64 * 2);
constexpr size_t OFF_WS2    = A256(OFF_WS1 + 64 * 128 * 2);
constexpr size_t OFF_WS3    = A256(OFF_WS2 + 64 * 64 * 2);          // 112x64 padded

// ---- WMMA helpers -----------------------------------------------------------
__device__ __forceinline__ v8f wmma32(v16h a, v16h b, v8f c) {
  return __builtin_amdgcn_wmma_f32_16x16x32_f16(false, a, false, b, (short)0, c,
                                                false, false);
}

// A-frag (16x32 f16) from row-major f32 matrix; lanes 0-15: M=lane, K=k0+{0..7,16..23}
// lanes 16-31: M=lane-16, K=k0+{8..15,24..31}
__device__ __forceinline__ v16h load_a_f32(const float* S, int rstride, int row0,
                                           int k0, int lane) {
  const float* p = S + (row0 + (lane & 15)) * rstride + k0 + ((lane >> 4) << 3);
  v16h a;
#pragma unroll
  for (int i = 0; i < 8; ++i) { a[i] = (_Float16)p[i]; a[8 + i] = (_Float16)p[16 + i]; }
  return a;
}

// B-frag (32x16 f16) for W^T tile from W stored row-major [N][krow] in f16.
// lane: N = n0 + lane%16, K = k0 + (lane<16 ? {0..7,16..23} : {8..15,24..31})
__device__ __forceinline__ v16h load_b_f16(const _Float16* W, int krow, int n0,
                                           int k0, int lane) {
  const _Float16* p = W + (n0 + (lane & 15)) * krow + k0 + ((lane >> 4) << 3);
  v16h b;
#pragma unroll
  for (int i = 0; i < 8; ++i) { b[i] = p[i]; b[8 + i] = p[16 + i]; }
  return b;
}

__device__ __forceinline__ float lrelu(float v) { return v >= 0.f ? v : SLOPE * v; }
__device__ __forceinline__ float sigm(float v) { return 1.f / (1.f + expf(-v)); }

// ---- preprocessing kernels --------------------------------------------------
__global__ void zero_i32(int* p, int n) {
  int i = blockIdx.x * blockDim.x + threadIdx.x;
  if (i < n) p[i] = 0;
}

__global__ void count_edges(const int* __restrict__ ei, int* __restrict__ cnt) {
  int e = blockIdx.x * blockDim.x + threadIdx.x;
  if (e < N_EDGES) atomicAdd(&cnt[ei[N_EDGES + e]], 1);
}

__global__ void scan_csr(const int* __restrict__ cnt, int* __restrict__ rowptr,
                         int* __restrict__ cursor) {
  __shared__ int part[256];
  int t = threadIdx.x;
  int b0 = t * 64;
  int s = 0;
  for (int i = 0; i < 64; ++i) s += cnt[b0 + i];
  part[t] = s;
  __syncthreads();
  for (int off = 1; off < 256; off <<= 1) {
    int v = (t >= off) ? part[t - off] : 0;
    __syncthreads();
    part[t] += v;
    __syncthreads();
  }
  int run = part[t] - s;  // exclusive prefix
  for (int i = 0; i < 64; ++i) {
    rowptr[b0 + i] = run;
    cursor[b0 + i] = run;
    run += cnt[b0 + i];
  }
  if (t == 255) rowptr[N_NODES] = run;
}

// meta: src_local_in_64block (6b) | attr0 (5b) | attr1 (5b); src shares dst's block
__global__ void scatter_edges(const int* __restrict__ ei, const int* __restrict__ ea,
                              int* __restrict__ cursor, unsigned* __restrict__ meta) {
  int e = blockIdx.x * blockDim.x + threadIdx.x;
  if (e < N_EDGES) {
    int src = ei[e], dst = ei[N_EDGES + e];
    int pos = atomicAdd(&cursor[dst], 1);
    meta[pos] = (unsigned)(src & 63) | ((unsigned)ea[2 * e] << 6) |
                ((unsigned)ea[2 * e + 1] << 11);
  }
}

__global__ void cvt_f16(const float* __restrict__ s, _Float16* __restrict__ d, int n) {
  int i = blockIdx.x * blockDim.x + threadIdx.x;
  if (i < n) d[i] = (_Float16)s[i];
}

__global__ void cvt_f16_pad(const float* __restrict__ s, _Float16* __restrict__ d,
                            int nsrc, int ntot) {
  int i = blockIdx.x * blockDim.x + threadIdx.x;
  if (i < ntot) d[i] = (i < nsrc) ? (_Float16)s[i] : (_Float16)0.f;
}

// ---- fused GNN: embeddings + b2e MLP + 6x(NNConv + GRU), all in LDS ---------
__global__ void __launch_bounds__(64) fused_gnn(
    const int* __restrict__ x_ids, const float* __restrict__ blockemb,
    const float* __restrict__ vec_data, const float* __restrict__ bondemb,
    const _Float16* __restrict__ w_b2e1, const float* __restrict__ b_b2e1,
    const _Float16* __restrict__ w_b2e2, const float* __restrict__ b_b2e2,
    const _Float16* __restrict__ w_root, const float* __restrict__ conv_bias,
    const _Float16* __restrict__ w_ih, const float* __restrict__ b_ih,
    const _Float16* __restrict__ w_hh, const float* __restrict__ b_hh,
    const int* __restrict__ rowptr, const unsigned* __restrict__ meta,
    float* __restrict__ node_h) {
  __shared__ float s_state[64][64];      // out == hstate
  __shared__ float s_agg[64][96];        // h0 (96 wide), then agg/m (64 wide)
  __shared__ float s_bond[20][64];
  __shared__ float s_edot[MAXE_WG];
  __shared__ unsigned s_meta[MAXE_WG];
  __shared__ int s_es[64];
  __shared__ int s_deg[64];
  __shared__ float s_inv[64];

  const int t = threadIdx.x;             // 64 threads = 2 waves
  const int lane = t & 31;
  const int base = blockIdx.x * 64;      // first node of this WG (4 graphs)
  const int mt0 = (t >> 5) * 2;          // wave's first 16-row M-tile

  for (int i = t; i < 20 * 64; i += 64) (&s_bond[0][0])[i] = bondemb[i];
  const int e0g = rowptr[base];
  {
    int a = rowptr[base + t], b = rowptr[base + t + 1];
    int es = a - e0g; if (es > MAXE_WG) es = MAXE_WG;
    int ee = b - e0g; if (ee > MAXE_WG) ee = MAXE_WG;
    s_es[t] = es; s_deg[t] = ee - es;
    int dg = b - a;
    s_inv[t] = 1.f / (float)(dg > 0 ? dg : 1);
  }
  int nE = rowptr[base + 64] - e0g; if (nE > MAXE_WG) nE = MAXE_WG;

  // h0 = concat(blockemb[x], vec[graph])
  {
    int id = x_ids[base + t];
#pragma unroll
    for (int f = 0; f < 64; ++f) s_agg[t][f] = blockemb[id * 64 + f];
    int g = (base + t) >> 4;
#pragma unroll
    for (int f = 0; f < 32; ++f) s_agg[t][64 + f] = vec_data[g * 32 + f];
  }
  __syncthreads();

  // b2e layer1: state = leaky(h0 @ w1^T + b1), K=96
#pragma unroll
  for (int mi = 0; mi < 2; ++mi) {
    int r0 = (mt0 + mi) * 16;
    v16h a0 = load_a_f32(&s_agg[0][0], 96, r0, 0, lane);
    v16h a1 = load_a_f32(&s_agg[0][0], 96, r0, 32, lane);
    v16h a2 = load_a_f32(&s_agg[0][0], 96, r0, 64, lane);
#pragma unroll
    for (int nt = 0; nt < 4; ++nt) {
      v8f c = {};
      c = wmma32(a0, load_b_f16(w_b2e1, 96, nt * 16, 0, lane), c);
      c = wmma32(a1, load_b_f16(w_b2e1, 96, nt * 16, 32, lane), c);
      c = wmma32(a2, load_b_f16(w_b2e1, 96, nt * 16, 64, lane), c);
      int col = nt * 16 + (lane & 15);
      float bb = b_b2e1[col];
      int rr = r0 + ((lane >> 4) << 3);
#pragma unroll
      for (int r = 0; r < 8; ++r) s_state[rr + r][col] = lrelu(c[r] + bb);
    }
  }
  __syncthreads();
  // b2e layer2: state = state @ w2^T + b2 (A preloaded before in-place writes)
#pragma unroll
  for (int mi = 0; mi < 2; ++mi) {
    int r0 = (mt0 + mi) * 16;
    v16h a0 = load_a_f32(&s_state[0][0], 64, r0, 0, lane);
    v16h a1 = load_a_f32(&s_state[0][0], 64, r0, 32, lane);
#pragma unroll
    for (int nt = 0; nt < 4; ++nt) {
      v8f c = {};
      c = wmma32(a0, load_b_f16(w_b2e2, 64, nt * 16, 0, lane), c);
      c = wmma32(a1, load_b_f16(w_b2e2, 64, nt * 16, 32, lane), c);
      int col = nt * 16 + (lane & 15);
      float bb = b_b2e2[col];
      int rr = r0 + ((lane >> 4) << 3);
#pragma unroll
      for (int r = 0; r < 8; ++r) s_state[rr + r][col] = c[r] + bb;
    }
  }
  __syncthreads();

  for (int step = 0; step < STEPS; ++step) {
    // E1: rank-1 edge message scalar part: dot(out[src], bond0[attr0])
    for (int e = t; e < nE; e += 64) {
      unsigned mm = meta[e0g + e];
      const float* sr = &s_state[mm & 63][0];
      const float* bd = &s_bond[(mm >> 6) & 31][0];
      float d = 0.f;
#pragma unroll
      for (int f = 0; f < 64; ++f) d += sr[f] * bd[f];
      s_edot[e] = d;
      s_meta[e] = mm;
    }
    __syncthreads();
    // E2: per-dst gather of dot*bond1, scatter-mean
    {
      float racc[64];
#pragma unroll
      for (int f = 0; f < 64; ++f) racc[f] = 0.f;
      int es = s_es[t], ee = es + s_deg[t];
      for (int e = es; e < ee; ++e) {
        float d = s_edot[e];
        const float* bd = &s_bond[(s_meta[e] >> 11) & 31][0];
#pragma unroll
        for (int f = 0; f < 64; ++f) racc[f] += d * bd[f];
      }
      float inv = s_inv[t];
#pragma unroll
      for (int f = 0; f < 64; ++f) s_agg[t][f] = racc[f] * inv;
    }
    __syncthreads();
    // M: m = leaky(agg + state @ root^T + bias)   (stored back into s_agg)
#pragma unroll
    for (int mi = 0; mi < 2; ++mi) {
      int r0 = (mt0 + mi) * 16;
      v16h a0 = load_a_f32(&s_state[0][0], 64, r0, 0, lane);
      v16h a1 = load_a_f32(&s_state[0][0], 64, r0, 32, lane);
#pragma unroll
      for (int nt = 0; nt < 4; ++nt) {
        v8f c = {};
        c = wmma32(a0, load_b_f16(w_root, 64, nt * 16, 0, lane), c);
        c = wmma32(a1, load_b_f16(w_root, 64, nt * 16, 32, lane), c);
        int col = nt * 16 + (lane & 15);
        float bb = conv_bias[col];
        int rr = r0 + ((lane >> 4) << 3);
#pragma unroll
        for (int r = 0; r < 8; ++r)
          s_agg[rr + r][col] = lrelu(c[r] + s_agg[rr + r][col] + bb);
      }
    }
    __syncthreads();
    // G: GRU (r,z,n gate order), h = (1-z)*n + z*h
#pragma unroll
    for (int mi = 0; mi < 2; ++mi) {
      int r0 = (mt0 + mi) * 16;
      v16h am0 = load_a_f32(&s_agg[0][0], 96, r0, 0, lane);
      v16h am1 = load_a_f32(&s_agg[0][0], 96, r0, 32, lane);
      v16h ah0 = load_a_f32(&s_state[0][0], 64, r0, 0, lane);
      v16h ah1 = load_a_f32(&s_state[0][0], 64, r0, 32, lane);
      v8f rfr[4], zfr[4];
#pragma unroll
      for (int j = 0; j < 4; ++j) {
        v8f cx = {}, ch = {};
        cx = wmma32(am0, load_b_f16(w_ih, 64, j * 16, 0, lane), cx);
        cx = wmma32(am1, load_b_f16(w_ih, 64, j * 16, 32, lane), cx);
        ch = wmma32(ah0, load_b_f16(w_hh, 64, j * 16, 0, lane), ch);
        ch = wmma32(ah1, load_b_f16(w_hh, 64, j * 16, 32, lane), ch);
        int col = j * 16 + (lane & 15);
        float br = b_ih[col] + b_hh[col];
        v8f rv;
#pragma unroll
        for (int r = 0; r < 8; ++r) rv[r] = sigm(cx[r] + ch[r] + br);
        rfr[j] = rv;
        v8f zx = {}, zh = {};
        zx = wmma32(am0, load_b_f16(w_ih, 64, 64 + j * 16, 0, lane), zx);
        zx = wmma32(am1, load_b_f16(w_ih, 64, 64 + j * 16, 32, lane), zx);
        zh = wmma32(ah0, load_b_f16(w_hh, 64, 64 + j * 16, 0, lane), zh);
        zh = wmma32(ah1, load_b_f16(w_hh, 64, 64 + j * 16, 32, lane), zh);
        float bz = b_ih[64 + col] + b_hh[64 + col];
        v8f zv;
#pragma unroll
        for (int r = 0; r < 8; ++r) zv[r] = sigm(zx[r] + zh[r] + bz);
        zfr[j] = zv;
      }
#pragma unroll
      for (int j = 0; j < 4; ++j) {
        v8f nx = {}, nh = {};
        nx = wmma32(am0, load_b_f16(w_ih, 64, 128 + j * 16, 0, lane), nx);
        nx = wmma32(am1, load_b_f16(w_ih, 64, 128 + j * 16, 32, lane), nx);
        nh = wmma32(ah0, load_b_f16(w_hh, 64, 128 + j * 16, 0, lane), nh);
        nh = wmma32(ah1, load_b_f16(w_hh, 64, 128 + j * 16, 32, lane), nh);
        int col = j * 16 + (lane & 15);
        float bxn = b_ih[128 + col], bhn = b_hh[128 + col];
        int rr = r0 + ((lane >> 4) << 3);
#pragma unroll
        for (int r = 0; r < 8; ++r) {
          float nn = tanhf((nx[r] + bxn) + rfr[j][r] * (nh[r] + bhn));
          float z = zfr[j][r];
          float hold = s_state[rr + r][col];
          s_state[rr + r][col] = (1.f - z) * nn + z * hold;
        }
      }
    }
    __syncthreads();
  }

#pragma unroll
  for (int f = 0; f < 64; ++f) node_h[(base + t) * 64 + f] = s_state[t][f];
}

// ---- stem head: [2048,128] -> 64 -> 64 -> 105, one wave per 16 stems --------
__global__ void __launch_bounds__(128) stem_head(
    const float* __restrict__ node_h, const float* __restrict__ stememb,
    const int* __restrict__ stemtype_ids, const int* __restrict__ stems_local,
    const int* __restrict__ stems_batch, const int* __restrict__ node_slices,
    const _Float16* __restrict__ w1, const float* __restrict__ b1,
    const _Float16* __restrict__ w2, const float* __restrict__ b2,
    const _Float16* __restrict__ w3, const float* __restrict__ b3,
    float* __restrict__ out) {
  __shared__ float sbuf[4][16][64];
  const int wave = threadIdx.x >> 5, lane = threadIdx.x & 31;
  const int s0 = (blockIdx.x * 4 + wave) * 16;
  const int srow = s0 + (lane & 15);
  const int nidx = node_slices[stems_batch[srow]] + stems_local[srow];
  const float* prow0 = node_h + (size_t)nidx * 64;
  const float* prow1 = stememb + (size_t)stemtype_ids[srow] * 64;

  v16h A[4];
#pragma unroll
  for (int kt = 0; kt < 4; ++kt) {
    int kb = kt * 32 + ((lane >> 4) << 3);
    const float* src = (kt < 2) ? (prow0 + kb) : (prow1 + (kb - 64));
    v16h a;
#pragma unroll
    for (int i = 0; i < 8; ++i) { a[i] = (_Float16)src[i]; a[8 + i] = (_Float16)src[16 + i]; }
    A[kt] = a;
  }
  const int rr = (lane >> 4) << 3;
  // layer1: leaky(sc @ w1^T + b1), K=128
#pragma unroll
  for (int nt = 0; nt < 4; ++nt) {
    v8f c = {};
#pragma unroll
    for (int kt = 0; kt < 4; ++kt)
      c = wmma32(A[kt], load_b_f16(w1, 128, nt * 16, kt * 32, lane), c);
    int col = nt * 16 + (lane & 15);
    float bb = b1[col];
#pragma unroll
    for (int r = 0; r < 8; ++r) sbuf[wave][rr + r][col] = lrelu(c[r] + bb);
  }
  __syncthreads();
  // layer2: leaky(s1 @ w2^T + b2)
  v16h B0 = load_a_f32(&sbuf[wave][0][0], 64, 0, 0, lane);
  v16h B1 = load_a_f32(&sbuf[wave][0][0], 64, 0, 32, lane);
  __syncthreads();
#pragma unroll
  for (int nt = 0; nt < 4; ++nt) {
    v8f c = {};
    c = wmma32(B0, load_b_f16(w2, 64, nt * 16, 0, lane), c);
    c = wmma32(B1, load_b_f16(w2, 64, nt * 16, 32, lane), c);
    int col = nt * 16 + (lane & 15);
    float bb = b2[col];
#pragma unroll
    for (int r = 0; r < 8; ++r) sbuf[wave][rr + r][col] = lrelu(c[r] + bb);
  }
  __syncthreads();
  // layer3: s2 @ w3^T + b3, N padded 105->112, write-guarded
  v16h C0 = load_a_f32(&sbuf[wave][0][0], 64, 0, 0, lane);
  v16h C1 = load_a_f32(&sbuf[wave][0][0], 64, 0, 32, lane);
#pragma unroll
  for (int nt = 0; nt < 7; ++nt) {
    v8f c = {};
    c = wmma32(C0, load_b_f16(w3, 64, nt * 16, 0, lane), c);
    c = wmma32(C1, load_b_f16(w3, 64, nt * 16, 32, lane), c);
    int col = nt * 16 + (lane & 15);
    if (col < OUT_PER_STEM) {
      float bb = b3[col];
#pragma unroll
      for (int r = 0; r < 8; ++r)
        out[(size_t)(s0 + rr + r) * OUT_PER_STEM + col] = c[r] + bb;
    }
  }
}

// ---- mol head: global mean pool -> 64 -> 1 ----------------------------------
__global__ void mol_head(const float* __restrict__ node_h,
                         const float* __restrict__ w1, const float* __restrict__ b1,
                         const float* __restrict__ w2, const float* __restrict__ b2,
                         float* __restrict__ out) {
  int g = blockIdx.x * blockDim.x + threadIdx.x;
  if (g >= N_GRAPHS) return;
  float gm[64];
#pragma unroll
  for (int f = 0; f < 64; ++f) gm[f] = 0.f;
  const float* p = node_h + (size_t)g * 16 * 64;
  for (int n = 0; n < 16; ++n)
#pragma unroll
    for (int f = 0; f < 64; ++f) gm[f] += p[n * 64 + f];
#pragma unroll
  for (int f = 0; f < 64; ++f) gm[f] *= (1.f / 16.f);
  float acc = b2[0];
  for (int k = 0; k < 64; ++k) {
    float v = b1[k];
#pragma unroll
    for (int f = 0; f < 64; ++f) v += gm[f] * w1[k * 64 + f];
    acc += lrelu(v) * w2[k];
  }
  out[g] = acc;
}

// ---------------------------------------------------------------------------
extern "C" void kernel_launch(void* const* d_in, const int* in_sizes, int n_in,
                              void* d_out, int out_size, void* d_ws, size_t ws_size,
                              hipStream_t stream) {
  (void)in_sizes; (void)n_in; (void)out_size; (void)ws_size;
  const int* x_ids        = (const int*)d_in[0];
  const int* stemtype_ids = (const int*)d_in[1];
  const int* edge_attr    = (const int*)d_in[2];
  const int* edge_index   = (const int*)d_in[3];
  const int* stems_local  = (const int*)d_in[5];
  const int* stems_batch  = (const int*)d_in[6];
  const int* node_slices  = (const int*)d_in[7];
  const float* vec_data   = (const float*)d_in[8];
  const float* blockemb   = (const float*)d_in[9];
  const float* stememb    = (const float*)d_in[10];
  const float* bondemb    = (const float*)d_in[11];
  const float* conv_root  = (const float*)d_in[12];
  const float* conv_bias  = (const float*)d_in[13];
  const float* b2e_w1 = (const float*)d_in[14]; const float* b2e_b1 = (const float*)d_in[15];
  const float* b2e_w2 = (const float*)d_in[16]; const float* b2e_b2 = (const float*)d_in[17];
  const float* gru_w_ih = (const float*)d_in[18]; const float* gru_b_ih = (const float*)d_in[19];
  const float* gru_w_hh = (const float*)d_in[20]; const float* gru_b_hh = (const float*)d_in[21];
  const float* s2p_w1 = (const float*)d_in[22]; const float* s2p_b1 = (const float*)d_in[23];
  const float* s2p_w2 = (const float*)d_in[24]; const float* s2p_b2 = (const float*)d_in[25];
  const float* s2p_w3 = (const float*)d_in[26]; const float* s2p_b3 = (const float*)d_in[27];
  const float* g2p_w1 = (const float*)d_in[28]; const float* g2p_b1 = (const float*)d_in[29];
  const float* g2p_w2 = (const float*)d_in[30]; const float* g2p_b2 = (const float*)d_in[31];

  char* ws = (char*)d_ws;
  float*     node_h = (float*)(ws + OFF_NODEH);
  int*       cnt    = (int*)(ws + OFF_CNT);
  int*       rowptr = (int*)(ws + OFF_ROWPTR);
  int*       cursor = (int*)(ws + OFF_CURSOR);
  unsigned*  meta   = (unsigned*)(ws + OFF_META);
  _Float16*  wroot  = (_Float16*)(ws + OFF_WROOT);
  _Float16*  wih    = (_Float16*)(ws + OFF_WIH);
  _Float16*  whh    = (_Float16*)(ws + OFF_WHH);
  _Float16*  wb1    = (_Float16*)(ws + OFF_WB1);
  _Float16*  wb2    = (_Float16*)(ws + OFF_WB2);
  _Float16*  w1h    = (_Float16*)(ws + OFF_WS1);
  _Float16*  w2h    = (_Float16*)(ws + OFF_WS2);
  _Float16*  w3h    = (_Float16*)(ws + OFF_WS3);

  // Build dst-node CSR for the edge messages (deterministic counts/offsets)
  zero_i32<<<N_NODES / 256, 256, 0, stream>>>(cnt, N_NODES);
  count_edges<<<N_EDGES / 256, 256, 0, stream>>>(edge_index, cnt);
  scan_csr<<<1, 256, 0, stream>>>(cnt, rowptr, cursor);
  scatter_edges<<<N_EDGES / 256, 256, 0, stream>>>(edge_index, edge_attr, cursor, meta);

  // f16 weight conversion (B-operand layout == row-major [N][K])
  cvt_f16<<<(64 * 64 + 255) / 256, 256, 0, stream>>>(conv_root, wroot, 64 * 64);
  cvt_f16<<<(192 * 64 + 255) / 256, 256, 0, stream>>>(gru_w_ih, wih, 192 * 64);
  cvt_f16<<<(192 * 64 + 255) / 256, 256, 0, stream>>>(gru_w_hh, whh, 192 * 64);
  cvt_f16<<<(64 * 96 + 255) / 256, 256, 0, stream>>>(b2e_w1, wb1, 64 * 96);
  cvt_f16<<<(64 * 64 + 255) / 256, 256, 0, stream>>>(b2e_w2, wb2, 64 * 64);
  cvt_f16<<<(64 * 128 + 255) / 256, 256, 0, stream>>>(s2p_w1, w1h, 64 * 128);
  cvt_f16<<<(64 * 64 + 255) / 256, 256, 0, stream>>>(s2p_w2, w2h, 64 * 64);
  cvt_f16_pad<<<(112 * 64 + 255) / 256, 256, 0, stream>>>(s2p_w3, w3h, 105 * 64, 112 * 64);

  // Fused message passing: 256 WGs x 64 threads (4 graphs each, LDS-resident)
  fused_gnn<<<N_NODES / 64, 64, 0, stream>>>(
      x_ids, blockemb, vec_data, bondemb, wb1, b2e_b1, wb2, b2e_b2, wroot,
      conv_bias, wih, gru_b_ih, whh, gru_b_hh, rowptr, meta, node_h);

  stem_head<<<N_STEMS / 64, 128, 0, stream>>>(
      node_h, stememb, stemtype_ids, stems_local, stems_batch, node_slices,
      w1h, s2p_b1, w2h, s2p_b2, w3h, s2p_b3, (float*)d_out);

  mol_head<<<N_GRAPHS / 256, 256, 0, stream>>>(
      node_h, g2p_w1, g2p_b1, g2p_w2, g2p_b2,
      (float*)d_out + (size_t)N_STEMS * OUT_PER_STEM);
}
